// WormholeAttention_56762287784305
// MI455X (gfx1250) — matrix operand order repaired and checked by reading, not compiled, not run on hardware
//
#include <hip/hip_runtime.h>
#include <math.h>

typedef __bf16 bf16;
typedef __attribute__((ext_vector_type(16))) __bf16 v16bf;
typedef __attribute__((ext_vector_type(8)))  float   v8f;

struct alignas(16) U4 { unsigned x, y, z, w; };
union FragAB { v16bf v; U4 u[2]; };
union FragC  { v8f   v; float f[8]; };

// ---- WMMA helpers (CDNA5 v_wmma_f32_16x16x32_bf16) ----
__device__ __forceinline__ v8f wmma_bf16(const FragAB& a, const FragAB& b, v8f c) {
  return __builtin_amdgcn_wmma_f32_16x16x32_bf16(false, a.v, false, b.v, (short)0, c, false, false);
}
// A fragment: lane<16 -> row lane, K 0..7 (bytes 0..15) and K 16..23 (bytes 32..47)
//             lane>=16 -> K 8..15 and K 24..31           (ISA 7.12.2, 16-bit A 16x32)
__device__ __forceinline__ void ld_frag_A(FragAB& f, const bf16* tile, int ldr, int lane) {
  const U4* p = (const U4*)(tile + (lane & 15) * ldr);
  int h = lane >> 4;
  f.u[0] = p[h];
  f.u[1] = p[h + 2];
}
// B fragment: lane<16 -> col lane, K 0..7 and K 16..23; lane>=16 -> K 8..15 and K 24..31
__device__ __forceinline__ void ld_frag_B(FragAB& f, const bf16* tile, int ldr, int lane) {
  const U4* p = (const U4*)(tile + (lane & 15) * ldr);
  int h = lane >> 4;
  f.u[0] = p[h];
  f.u[1] = p[h + 2];
}

// ---------------- conversion kernels ----------------
__global__ void k_cvt(const float* __restrict__ x, bf16* __restrict__ y, long n) {
  long i = (long)blockIdx.x * blockDim.x + threadIdx.x;
  long st = (long)gridDim.x * blockDim.x;
  for (; i < n; i += st) y[i] = (bf16)x[i];
}
// W[rowoff+k][n] (row-stride dout) -> Wt[n][k]  (bf16, K-contiguous)
__global__ void k_cvt_tr(const float* __restrict__ W, bf16* __restrict__ Wt,
                         int din, int dout, int rowoff) {
  long n = (long)din * dout;
  long i = (long)blockIdx.x * blockDim.x + threadIdx.x;
  long st = (long)gridDim.x * blockDim.x;
  for (; i < n; i += st) {
    long nn = i / din, kk = i - nn * din;
    Wt[i] = (bf16)W[(rowoff + kk) * (long)dout + nn];
  }
}
__global__ void k_copy(const float* __restrict__ x, float* __restrict__ y, long n) {
  long i = (long)blockIdx.x * blockDim.x + threadIdx.x;
  long st = (long)gridDim.x * blockDim.x;
  for (; i < n; i += st) y[i] = x[i];
}

// ---------------- generic WMMA GEMM ----------------
// C[M,N] = A[M,K] @ Wt[N,K]^T (+bias) ; tile 128x64, K-step 32, 256 thr / 8 waves
// wave w: rows (w&3)*32..+32, cols (w>>2)*32..+32 -> 2 A-frags + 2 B-frags -> 4 wmma
#define BM 128
#define BN 64
#define BKK 32
#define LDA 48
#define LDB 48

__global__ __launch_bounds__(256, 1) void k_gemm(
    const bf16* __restrict__ A, const bf16* __restrict__ Bw,
    const float* __restrict__ bias, int M, int N, int K, int mode,
    bf16* __restrict__ outB, float* __restrict__ outF,
    const float* __restrict__ addend, const float* __restrict__ coF) {
  __shared__ __align__(16) bf16 As[BM * LDA];
  __shared__ __align__(16) bf16 Bs[BN * LDB];
  int tid = threadIdx.x, lane = tid & 31, w = tid >> 5;
  int m0 = blockIdx.y * BM, n0 = blockIdx.x * BN;
  int ms = (w & 3) * 32;      // 32-row strip (two 16-row substrips)
  int ng = (w >> 2) * 32;     // 32-col group (two 16-col tiles)

  FragC c[2][2];
  for (int s = 0; s < 2; s++)
    for (int t = 0; t < 2; t++)
      for (int e = 0; e < 8; e++) c[s][t].f[e] = 0.f;

  for (int k0 = 0; k0 < K; k0 += BKK) {
    for (int it = 0; it < 2; it++) { // A tile 128x32 bf16 : two 16B chunks / thread
      int r = (tid >> 2) + it * 64, q = tid & 3;
      *(U4*)(As + r * LDA + q * 8) = *(const U4*)(A + (long)(m0 + r) * K + k0 + q * 8);
    }
    { // B tile 64x32 bf16 : one 16B chunk / thread
      int r = tid >> 2, q = tid & 3;
      *(U4*)(Bs + r * LDB + q * 8) = *(const U4*)(Bw + (long)(n0 + r) * K + k0 + q * 8);
    }
    __syncthreads();
    FragAB a0, a1, b0, b1;
    ld_frag_A(a0, As + ms * LDA, LDA, lane);
    ld_frag_A(a1, As + (ms + 16) * LDA, LDA, lane);
    ld_frag_B(b0, Bs + ng * LDB, LDB, lane);
    ld_frag_B(b1, Bs + (ng + 16) * LDB, LDB, lane);
    c[0][0].v = wmma_bf16(a0, b0, c[0][0].v);
    c[0][1].v = wmma_bf16(a0, b1, c[0][1].v);
    c[1][0].v = wmma_bf16(a1, b0, c[1][0].v);
    c[1][1].v = wmma_bf16(a1, b1, c[1][1].v);
    __syncthreads();
  }

  int hi = lane >> 4, ln = lane & 15;
  for (int s = 0; s < 2; s++) {
    for (int t = 0; t < 2; t++) {
      int n = n0 + ng + t * 16 + ln;
      float bv = bias ? bias[n] : 0.f;
      for (int r = 0; r < 8; r++) {
        long m = m0 + ms + s * 16 + r + 8 * hi;
        float v = c[s][t].f[r] + bv;
        long idx = m * (long)N + n;
        if (mode == 0) {                      // bf16 row-major (q, k)
          outB[idx] = (bf16)v;
        } else if (mode == 1) {               // v, stored transposed per batch: vT[b][n][s]
          long bb = m >> 11, sloc = m & 2047;
          outB[(bb * (long)N + n) * 2048 + sloc] = (bf16)v;
        } else if (mode == 2) {               // f32 (gtop = band@Wg_top + bg)
          outF[idx] = v;
        } else if (mode == 3) {               // co_proj: f32 + bf16
          outF[idx] = v;
          outB[idx] = (bf16)v;
        } else {                              // gate + accumulate
          float g = v + addend[idx];
          g = 1.f / (1.f + __expf(-g));
          g = (g > 0.5f) ? g : 0.f;
          outF[idx] += g * coF[idx] * (1.f / 3.f);
        }
      }
    }
  }
}

// ---------------- flash cross-attention ----------------
// co[b, q0..q0+32, 0..512] = softmax(Q Kt / sqrt(256)) @ V, online softmax, BK=64
// 8 waves: scores -> each wave one 16x16 tile (2 q-strips x 4 kv-tiles);
// PV -> wave w: q-strip (w&1), dv quarter (w>>1), K=64 (2 wmma per dv tile)
#define FQ 32
#define FK 64
#define DQK 256
#define DV 512
#define LQ 272
#define LV 80
#define LP 80
#define LSC 68

__global__ __launch_bounds__(256, 1) void k_flash(
    const bf16* __restrict__ Q, const bf16* __restrict__ Kb,
    const bf16* __restrict__ Vt, bf16* __restrict__ O, float scale) {
  extern __shared__ char smem[];
  bf16* qs = (bf16*)smem;                 // 32 x 272
  bf16* ks = qs + FQ * LQ;                // 64 x 272
  bf16* vs = ks + FK * LQ;                // 512 x 80
  float* sc = (float*)(vs + DV * LV);     // 32 x 68
  bf16* ps = (bf16*)(sc + FQ * LSC);      // 32 x 80
  float* mrow = (float*)(ps + FQ * LP);
  float* lrow = mrow + FQ;
  float* crow = lrow + FQ;

  int tid = threadIdx.x, lane = tid & 31, w = tid >> 5;
  int hi = lane >> 4, ln = lane & 15;
  int q0 = blockIdx.x * FQ, b = blockIdx.y;

  long qbase = ((long)b * 2048 + q0) * DQK;
  for (int idx = tid; idx < FQ * DQK / 8; idx += 256) {
    int r = idx >> 5, q = idx & 31;
    *(U4*)(qs + r * LQ + q * 8) = *(const U4*)(Q + qbase + (long)r * DQK + q * 8);
  }
  if (tid < FQ) { mrow[tid] = -1e30f; lrow[tid] = 0.f; }
  __syncthreads();

  int sw = w & 1;   // q strip (16 rows)
  int qd = w >> 1;  // dv quarter (128 cols = 8 tiles)
  FragC acc[8];
  for (int t = 0; t < 8; t++)
    for (int e = 0; e < 8; e++) acc[t].f[e] = 0.f;

  for (int kv0 = 0; kv0 < 2048; kv0 += FK) {
    long kbase = ((long)b * 2048 + kv0) * DQK;
    for (int idx = tid; idx < FK * DQK / 8; idx += 256) {
      int r = idx >> 5, q = idx & 31;
      *(U4*)(ks + r * LQ + q * 8) = *(const U4*)(Kb + kbase + (long)r * DQK + q * 8);
    }
    for (int idx = tid; idx < DV * FK / 8; idx += 256) {
      int r = idx >> 3, q = idx & 7;
      *(U4*)(vs + r * LV + q * 8) =
          *(const U4*)(Vt + ((long)b * DV + r) * 2048 + kv0 + q * 8);
    }
    __syncthreads();

    { // scores: every wave computes one 16x16 tile, K=256
      int qt = (w & 1) * 16, kt = (w >> 1) * 16;
      FragC s;
      for (int e = 0; e < 8; e++) s.f[e] = 0.f;
      for (int kk = 0; kk < DQK; kk += 32) {
        FragAB a; ld_frag_A(a, qs + qt * LQ + kk, LQ, lane);
        FragAB bb; ld_frag_B(bb, ks + kt * LQ + kk, LQ, lane);
        s.v = wmma_bf16(a, bb, s.v);
      }
      for (int r = 0; r < 8; r++)
        sc[(qt + r + 8 * hi) * LSC + kt + ln] = s.f[r] * scale;
    }
    __syncthreads();

    if (tid < FQ) {  // online softmax per q-row (wave 0, full EXEC)
      int r = tid;
      float mo = mrow[r], mx = mo;
      for (int c2 = 0; c2 < FK; c2++) mx = fmaxf(mx, sc[r * LSC + c2]);
      float corr = __expf(mo - mx);
      float sum = 0.f;
      for (int c2 = 0; c2 < FK; c2++) {
        float p = __expf(sc[r * LSC + c2] - mx);
        ps[r * LP + c2] = (bf16)p;
        sum += p;
      }
      mrow[r] = mx; lrow[r] = lrow[r] * corr + sum; crow[r] = corr;
    }
    __syncthreads();

    float cf[8];
    for (int r = 0; r < 8; r++) cf[r] = crow[sw * 16 + r + 8 * hi];
    FragAB a0, a1;
    ld_frag_A(a0, ps + sw * 16 * LP, LP, lane);
    ld_frag_A(a1, ps + sw * 16 * LP + 32, LP, lane);
    for (int t = 0; t < 8; t++) {
      for (int e = 0; e < 8; e++) acc[t].f[e] *= cf[e];
      FragAB b0, b1;
      ld_frag_B(b0, vs + (qd * 128 + t * 16) * LV, LV, lane);
      ld_frag_B(b1, vs + (qd * 128 + t * 16) * LV + 32, LV, lane);
      acc[t].v = wmma_bf16(a0, b0, acc[t].v);
      acc[t].v = wmma_bf16(a1, b1, acc[t].v);
    }
    __syncthreads();
  }

  float lv[8];
  for (int r = 0; r < 8; r++) lv[r] = 1.f / lrow[sw * 16 + r + 8 * hi];
  for (int t = 0; t < 8; t++) {
    int n = qd * 128 + t * 16 + ln;
    for (int r = 0; r < 8; r++) {
      long row = (long)b * 2048 + q0 + sw * 16 + r + 8 * hi;
      O[row * (long)DV + n] = (bf16)(acc[t].f[r] * lv[r]);
    }
  }
}

// ---------------- host orchestration ----------------
extern "C" void kernel_launch(void* const* d_in, const int* in_sizes, int n_in,
                              void* d_out, int out_size, void* d_ws, size_t ws_size,
                              hipStream_t stream) {
  (void)in_sizes; (void)n_in; (void)out_size; (void)ws_size;
  const int d[4] = {512, 768, 1024, 2048};
  const long M = 4L * 2048;  // flattened B*S
  // pytree order: band0..band3, then params keys sorted: g,k,o,q,v; each [ (W,b) x4 ]
  const int IG = 4, IK = 12, IO = 20, IQ = 28, IV = 36;
  auto Wp = [&](int base, int i) { return (const float*)d_in[base + 2 * i]; };
  auto Bp = [&](int base, int i) { return (const float*)d_in[base + 2 * i + 1]; };

  char* ws = (char*)d_ws;
  long off = 0;
  auto alloc = [&](long bytes) { long o = off; off += (bytes + 255) & ~255L; return o; };
  long o_band[4], o_wq[4], o_wk[4], o_wv[4], o_wo[4], o_wgt[4], o_wgb[4];
  long o_q[4], o_k[4], o_vT[4];
  for (int i = 0; i < 4; i++) o_band[i] = alloc(M * d[i] * 2);
  for (int i = 0; i < 4; i++) o_wq[i] = alloc((long)d[i] * 256 * 2);
  for (int i = 0; i < 4; i++) o_wk[i] = alloc((long)d[i] * 256 * 2);
  for (int i = 0; i < 4; i++) o_wv[i] = alloc((long)d[i] * 512 * 2);
  for (int i = 0; i < 4; i++) o_wo[i] = alloc((long)d[i] * 512 * 2);
  for (int i = 0; i < 4; i++) o_wgt[i] = alloc((long)d[i] * d[i] * 2);
  for (int i = 0; i < 4; i++) o_wgb[i] = alloc((long)d[i] * d[i] * 2);
  for (int i = 0; i < 4; i++) o_q[i] = alloc(M * 256 * 2);
  for (int i = 0; i < 4; i++) o_k[i] = alloc(M * 256 * 2);
  for (int i = 0; i < 4; i++) o_vT[i] = alloc(M * 512 * 2);
  long o_gtop = alloc(M * 2048 * 4);
  long o_co   = alloc(M * 512 * 2);
  long o_cof  = alloc(M * 2048 * 4);
  long o_cob  = alloc(M * 2048 * 2);

  long outOff[4]; { long a = 0; for (int i = 0; i < 4; i++) { outOff[i] = a; a += M * d[i]; } }

  // convert bands + weights to bf16 (weights transposed to [dout][din])
  for (int i = 0; i < 4; i++)
    k_cvt<<<1024, 256, 0, stream>>>((const float*)d_in[i], (bf16*)(ws + o_band[i]), M * d[i]);
  for (int i = 0; i < 4; i++) {
    k_cvt_tr<<<512, 256, 0, stream>>>(Wp(IQ, i), (bf16*)(ws + o_wq[i]), d[i], 256, 0);
    k_cvt_tr<<<512, 256, 0, stream>>>(Wp(IK, i), (bf16*)(ws + o_wk[i]), d[i], 256, 0);
    k_cvt_tr<<<512, 256, 0, stream>>>(Wp(IV, i), (bf16*)(ws + o_wv[i]), d[i], 512, 0);
    k_cvt_tr<<<512, 256, 0, stream>>>(Wp(IO, i), (bf16*)(ws + o_wo[i]), 512, d[i], 0);
    k_cvt_tr<<<512, 256, 0, stream>>>(Wp(IG, i), (bf16*)(ws + o_wgt[i]), d[i], d[i], 0);
    k_cvt_tr<<<512, 256, 0, stream>>>(Wp(IG, i), (bf16*)(ws + o_wgb[i]), d[i], d[i], d[i]);
  }

  auto gemm = [&](long oA, long oB, const float* bias, int N, int K, int mode,
                  bf16* outB, float* outF, const float* add, const float* coF) {
    dim3 g((unsigned)(N / BN), (unsigned)(M / BM));
    k_gemm<<<g, 256, 0, stream>>>((const bf16*)(ws + oA), (const bf16*)(ws + oB),
                                  bias, (int)M, N, K, mode, outB, outF, add, coF);
  };

  // projections (k/v once per band; q once per band)
  for (int i = 0; i < 4; i++) {
    gemm(o_band[i], o_wq[i], Bp(IQ, i), 256, d[i], 0, (bf16*)(ws + o_q[i]), nullptr, nullptr, nullptr);
    gemm(o_band[i], o_wk[i], Bp(IK, i), 256, d[i], 0, (bf16*)(ws + o_k[i]), nullptr, nullptr, nullptr);
    gemm(o_band[i], o_wv[i], Bp(IV, i), 512, d[i], 1, (bf16*)(ws + o_vT[i]), nullptr, nullptr, nullptr);
  }

  const long flashLds = (long)FQ * LQ * 2 + (long)FK * LQ * 2 + (long)DV * LV * 2 +
                        (long)FQ * LSC * 4 + (long)FQ * LP * 2 + 3L * FQ * 4;
  const float scale = 1.0f / 16.0f;  // 1/sqrt(256)

  for (int i = 0; i < 4; i++) {
    // gtop = band_i @ Wg_top + bg (reused across j)
    gemm(o_band[i], o_wgt[i], Bp(IG, i), d[i], d[i], 2, nullptr, (float*)(ws + o_gtop), nullptr, nullptr);
    // out_i = band_i
    k_copy<<<1024, 256, 0, stream>>>((const float*)d_in[i], (float*)d_out + outOff[i], M * d[i]);
    for (int j = 0; j < 4; j++) {
      if (j == i) continue;
      k_flash<<<dim3(2048 / FQ, 4), 256, flashLds, stream>>>(
          (const bf16*)(ws + o_q[i]), (const bf16*)(ws + o_k[j]),
          (const bf16*)(ws + o_vT[j]), (bf16*)(ws + o_co), scale);
      // co_proj = co @ Wo + bo  -> f32 + bf16
      gemm(o_co, o_wo[i], Bp(IO, i), d[i], 512, 3,
           (bf16*)(ws + o_cob), (float*)(ws + o_cof), nullptr, nullptr);
      // gate = sigmoid(co_proj @ Wg_bot + gtop); threshold; out += g*co_proj/3
      gemm(o_cob, o_wgb[i], nullptr, d[i], d[i], 4,
           nullptr, (float*)d_out + outOff[i], (const float*)(ws + o_gtop),
           (const float*)(ws + o_cof));
    }
  }
}
